// SparseConvCausalAttention_51986284151357
// MI455X (gfx1250) — compile-verified, matrix operand order checked by reading, not run on hardware
//
#include <hip/hip_runtime.h>
#include <hip/hip_bf16.h>

// ---------------- problem constants ----------------
constexpr int B_    = 8;
constexpr int SEQ_  = 1279;
constexpr int SEQP_ = 1280;
constexpr int DIM_  = 512;
constexpr int DH_   = 64;
constexpr int NH_   = 64;           // 8 batches * 8 heads
constexpr int TXT_  = 256;          // text length
constexpr int WIN_  = 25;           // 5x5 window
constexpr float SCALE_F = 0.125f;   // 64^-0.5
#define NEG_F (-3.402823466e38f)

// ---------------- WMMA plumbing ----------------
typedef __attribute__((ext_vector_type(16))) __bf16 v16bf;
typedef __attribute__((ext_vector_type(8)))  float  f32x8;

union Frag { v16bf v; unsigned u[8]; };

__device__ __forceinline__ f32x8 zero8() {
  f32x8 z = {0.f,0.f,0.f,0.f,0.f,0.f,0.f,0.f};
  return z;
}

// 16-bit A/B fragment K mapping (ISA 7.12.2): lane L, VGPR vi holds K pair at
// (vi<4 ? 0:16) + 8*(L/16) + 2*(vi&3)
__device__ __forceinline__ int kmap(int vi, int g) {
  return ((vi < 4) ? 0 : 16) + g * 8 + (vi & 3) * 2;
}

// Load an A-layout (or B-from-transposed-source) fragment: rowptr points at
// (row = lane%16) of a row-major bf16 matrix, 4-byte aligned.
__device__ __forceinline__ void load_frag_rows(Frag& f, const unsigned short* rowptr, int g) {
  const unsigned* p32 = reinterpret_cast<const unsigned*>(rowptr);
#pragma unroll
  for (int vi = 0; vi < 8; ++vi) f.u[vi] = p32[kmap(vi, g) >> 1];
}

__device__ __forceinline__ f32x8 wmma_bf16(const Frag& a, const Frag& b, f32x8 c) {
  return __builtin_amdgcn_wmma_f32_16x16x32_bf16(false, a.v, false, b.v,
                                                 (short)0, c, false, false);
}

__device__ __forceinline__ unsigned short f2bf(float f) {
  __hip_bfloat16 h = __float2bfloat16(f);
  return reinterpret_cast<unsigned short&>(h);
}
__device__ __forceinline__ float bf2f(unsigned u) {
  unsigned short s = (unsigned short)u;
  __hip_bfloat16 h = reinterpret_cast<__hip_bfloat16&>(s);
  return __bfloat162float(h);
}

// ---------------- conversion kernels ----------------
__global__ void k_cvt_x(const float* __restrict__ x, unsigned short* __restrict__ xb) {
  int idx = blockIdx.x * 256 + threadIdx.x;
  if (idx >= B_ * SEQP_ * DIM_) return;
  int c = idx % DIM_;
  int row = idx / DIM_;
  int t = row % SEQP_, b = row / SEQP_;
  float v = (t < SEQ_) ? x[((size_t)(b * SEQ_ + t)) * DIM_ + c] : 0.f;
  xb[idx] = f2bf(v);
}

__global__ void k_cvt_wqkv(const float* __restrict__ w, unsigned short* __restrict__ wt) {
  int idx = blockIdx.x * 256 + threadIdx.x;            // over 1536*512 (transposed)
  if (idx >= 1536 * 512) return;
  int k = idx % 512, n = idx / 512;
  wt[idx] = f2bf(w[(size_t)k * 1536 + n]);
}

__global__ void k_cvt_wout(const float* __restrict__ w, unsigned short* __restrict__ wt) {
  int idx = blockIdx.x * 256 + threadIdx.x;            // over 512*512 (transposed)
  if (idx >= 512 * 512) return;
  int k = idx % 512, n = idx / 512;
  wt[idx] = f2bf(w[(size_t)k * 512 + n]);
}

// ---------------- QKV GEMM: xp(8,1280,512) @ w_qkv(512,1536) ----------------
// 32x64 output per wave (2 M x 4 N subtiles) -> 8 WMMAs per 32-k step with
// 2 A-frag + 4 B-frag loads. Writes head-major q/k/v (NH,1280,64) bf16
// (q pre-scaled) plus v_text transposed (NH,64,256) for the P.V WMMA B operand.
__global__ void __launch_bounds__(128) k_qkv(const unsigned short* __restrict__ xb,
                                             const unsigned short* __restrict__ wqkvT,
                                             unsigned short* __restrict__ qb,
                                             unsigned short* __restrict__ kb,
                                             unsigned short* __restrict__ vb,
                                             unsigned short* __restrict__ vtt) {
  int wid = threadIdx.x >> 5, lane = threadIdx.x & 31;
  int w = blockIdx.x * 4 + wid;          // 8 * 40 * 24 waves
  int b = w / (40 * 24);
  int rem = w % (40 * 24);
  int mtg = rem / 24, ntg = rem % 24;    // 32-row M group, 64-col N group
  int r = lane & 15, g = lane >> 4;

  const unsigned short* arow0 = xb + ((size_t)(b * SEQP_ + mtg * 32 + r)) * DIM_;
  const unsigned short* arow1 = arow0 + (size_t)16 * DIM_;
  const unsigned short* brow[4];
#pragma unroll
  for (int t = 0; t < 4; ++t)
    brow[t] = wqkvT + ((size_t)(ntg * 64 + t * 16 + r)) * DIM_;

  f32x8 acc[2][4];
#pragma unroll
  for (int mi = 0; mi < 2; ++mi)
#pragma unroll
    for (int t = 0; t < 4; ++t) acc[mi][t] = zero8();

  for (int k0 = 0; k0 < DIM_; k0 += 32) {
    Frag a0, a1;
    load_frag_rows(a0, arow0 + k0, g);
    load_frag_rows(a1, arow1 + k0, g);
#pragma unroll
    for (int t = 0; t < 4; ++t) {
      Frag bf;
      load_frag_rows(bf, brow[t] + k0, g);
      acc[0][t] = wmma_bf16(a0, bf, acc[0][t]);
      acc[1][t] = wmma_bf16(a1, bf, acc[1][t]);
    }
  }

  int which = ntg / 8;                   // 64-col group never straddles q/k/v
  int h = ntg % 8;
  float sc = (which == 0) ? SCALE_F : 1.f;
  unsigned short* dst = (which == 0) ? qb : ((which == 1) ? kb : vb);
  size_t hb = (size_t)(b * 8 + h) * SEQP_;
#pragma unroll
  for (int mi = 0; mi < 2; ++mi)
#pragma unroll
    for (int t = 0; t < 4; ++t) {
      int d = t * 16 + r;                // C-layout: n = lane%16
#pragma unroll
      for (int s = 0; s < 8; ++s) {
        int trow = mtg * 32 + mi * 16 + s + 8 * g;   // m = slot + 8*(lane/16)
        unsigned short bv = f2bf(acc[mi][t][s] * sc);
        dst[(hb + trow) * DH_ + d] = bv;
        if (which == 2 && trow < TXT_)
          vtt[((size_t)(b * 8 + h) * DH_ + d) * TXT_ + trow] = bv;
      }
    }
}

// ---------------- text causal self-attention ----------------
// One wave = 16 queries. LDS: per-wave 16x256 f32 score rows.
__global__ void __launch_bounds__(128) k_text(const unsigned short* __restrict__ qb,
                                              const unsigned short* __restrict__ kb,
                                              const unsigned short* __restrict__ vtt,
                                              unsigned short* __restrict__ ao) {
  extern __shared__ float smem[];
  int wid = threadIdx.x >> 5, lane = threadIdx.x & 31;
  int w = blockIdx.x * 4 + wid;
  int head = w >> 4;
  int qbase = (w & 15) << 4;
  int r = lane & 15, g = lane >> 4;
  float* S = smem + (size_t)wid * 16 * 256;

  const unsigned short* qrow = qb + ((size_t)head * SEQP_ + qbase + r) * DH_;
  Frag a0, a1;
  load_frag_rows(a0, qrow, g);
  load_frag_rows(a1, qrow + 32, g);

  // scores + causal mask -> LDS
  for (int jt = 0; jt < 16; ++jt) {
    const unsigned short* krow = kb + ((size_t)head * SEQP_ + jt * 16 + r) * DH_;
    Frag b0, b1;
    load_frag_rows(b0, krow, g);
    load_frag_rows(b1, krow + 32, g);
    f32x8 c = zero8();
    c = wmma_bf16(a0, b0, c);
    c = wmma_bf16(a1, b1, c);
    int kj = jt * 16 + r;
#pragma unroll
    for (int s = 0; s < 8; ++s) {
      int m = s + 8 * g, qi = qbase + m;
      S[m * 256 + kj] = (kj > qi) ? NEG_F : c[s];
    }
  }
  __syncthreads();

  // softmax (2 lanes per row)
  {
    int row = lane >> 1, sub = lane & 1;
    float mx = NEG_F;
    for (int c = sub; c < 256; c += 2) mx = fmaxf(mx, S[row * 256 + c]);
    mx = fmaxf(mx, __shfl_xor(mx, 1));
    float sm = 0.f;
    for (int c = sub; c < 256; c += 2) {
      float e = __expf(S[row * 256 + c] - mx);
      S[row * 256 + c] = e;
      sm += e;
    }
    sm += __shfl_xor(sm, 1);
    float inv = 1.f / sm;
    for (int c = sub; c < 256; c += 2) S[row * 256 + c] *= inv;
  }
  __syncthreads();

  // O = P.V  (B from v_text^T rows)
  f32x8 o0 = zero8(), o1 = zero8(), o2 = zero8(), o3 = zero8();
  f32x8* o[4] = {&o0, &o1, &o2, &o3};
  for (int j2 = 0; j2 < 8; ++j2) {
    Frag p;
#pragma unroll
    for (int vi = 0; vi < 8; ++vi) {
      int k = j2 * 32 + kmap(vi, g);
      unsigned lo = f2bf(S[r * 256 + k]);
      unsigned hi = f2bf(S[r * 256 + k + 1]);
      p.u[vi] = lo | (hi << 16);
    }
#pragma unroll
    for (int t = 0; t < 4; ++t) {
      const unsigned short* vrow = vtt + ((size_t)head * DH_ + t * 16 + r) * TXT_ + j2 * 32;
      Frag bf;
      load_frag_rows(bf, vrow, g);
      *o[t] = wmma_bf16(p, bf, *o[t]);
    }
  }
#pragma unroll
  for (int t = 0; t < 4; ++t)
#pragma unroll
    for (int s = 0; s < 8; ++s) {
      int m = s + 8 * g;
      ao[((size_t)head * SEQP_ + qbase + m) * DH_ + t * 16 + r] = f2bf((*o[t])[s]);
    }
}

// ---------------- image attention: 256 text cols (WMMA) + 25 window cols (VALU) ----
__global__ void __launch_bounds__(128) k_img(const unsigned short* __restrict__ qb,
                                             const unsigned short* __restrict__ kb,
                                             const unsigned short* __restrict__ vb,
                                             const unsigned short* __restrict__ vtt,
                                             const unsigned char* __restrict__ maskp,
                                             unsigned short* __restrict__ ao) {
  extern __shared__ float smem[];
  const int SS = 288;                         // padded score row stride
  int wid = threadIdx.x >> 5, lane = threadIdx.x & 31;
  int w = blockIdx.x * 4 + wid;
  int head = w >> 6;
  int qbase = (w & 63) << 4;                  // image-query tile base (0..1023)
  int b = head >> 3;
  int r = lane & 15, g = lane >> 4;
  float* S = smem + (size_t)wid * (16 * SS + 16 * 64);
  float* O = S + 16 * SS;

  const unsigned short* qrow = qb + ((size_t)head * SEQP_ + TXT_ + qbase + r) * DH_;
  Frag a0, a1;
  load_frag_rows(a0, qrow, g);
  load_frag_rows(a1, qrow + 32, g);

  // text scores (with text pad mask)
  for (int jt = 0; jt < 16; ++jt) {
    const unsigned short* krow = kb + ((size_t)head * SEQP_ + jt * 16 + r) * DH_;
    Frag b0, b1;
    load_frag_rows(b0, krow, g);
    load_frag_rows(b1, krow + 32, g);
    f32x8 c = zero8();
    c = wmma_bf16(a0, b0, c);
    c = wmma_bf16(a1, b1, c);
    int kj = jt * 16 + r;
    bool pad = (maskp[b * TXT_ + kj] == 0);
#pragma unroll
    for (int s = 0; s < 8; ++s) {
      int m = s + 8 * g;
      S[m * SS + kj] = pad ? NEG_F : c[s];
    }
  }

  // window scores (causal conv mask); lane owns query r, g splits the 25 taps
  {
    int qi = qbase + r, rr = qi >> 5, cc = qi & 31;
    const unsigned short* qp = qb + ((size_t)head * SEQP_ + TXT_ + qi) * DH_;
    int j0 = g ? 13 : 0, j1 = g ? 25 : 13;
    for (int j = j0; j < j1; ++j) {
      int kr = rr + j / 5 - 2, kc = cc + j % 5 - 2;
      int kidx = kr * 32 + kc;
      float s = NEG_F;
      if (kr >= 0 && kr < 32 && kc >= 0 && kc < 32 && kidx <= qi) {
        const unsigned short* kp = kb + ((size_t)head * SEQP_ + TXT_ + kidx) * DH_;
        float acc = 0.f;
        const unsigned* q32 = (const unsigned*)qp;
        const unsigned* k32 = (const unsigned*)kp;
#pragma unroll 8
        for (int d = 0; d < DH_ / 2; ++d) {
          unsigned uq = q32[d], uk = k32[d];
          acc += bf2f(uq & 0xffff) * bf2f(uk & 0xffff) + bf2f(uq >> 16) * bf2f(uk >> 16);
        }
        s = acc;
      }
      S[r * SS + 256 + j] = s;
    }
  }
  __syncthreads();

  // softmax over 281 columns
  {
    int row = lane >> 1, sub = lane & 1;
    float mx = NEG_F;
    for (int c = sub; c < 281; c += 2) mx = fmaxf(mx, S[row * SS + c]);
    mx = fmaxf(mx, __shfl_xor(mx, 1));
    float sm = 0.f;
    for (int c = sub; c < 281; c += 2) {
      float e = __expf(S[row * SS + c] - mx);
      S[row * SS + c] = e;
      sm += e;
    }
    sm += __shfl_xor(sm, 1);
    float inv = 1.f / sm;
    for (int c = sub; c < 281; c += 2) S[row * SS + c] *= inv;
  }
  __syncthreads();

  // text part: O = P.v_text via WMMA -> LDS out tile
  f32x8 o0 = zero8(), o1 = zero8(), o2 = zero8(), o3 = zero8();
  f32x8* o[4] = {&o0, &o1, &o2, &o3};
  for (int j2 = 0; j2 < 8; ++j2) {
    Frag p;
#pragma unroll
    for (int vi = 0; vi < 8; ++vi) {
      int k = j2 * 32 + kmap(vi, g);
      unsigned lo = f2bf(S[r * SS + k]);
      unsigned hi = f2bf(S[r * SS + k + 1]);
      p.u[vi] = lo | (hi << 16);
    }
#pragma unroll
    for (int t = 0; t < 4; ++t) {
      const unsigned short* vrow = vtt + ((size_t)head * DH_ + t * 16 + r) * TXT_ + j2 * 32;
      Frag bf;
      load_frag_rows(bf, vrow, g);
      *o[t] = wmma_bf16(p, bf, *o[t]);
    }
  }
#pragma unroll
  for (int t = 0; t < 4; ++t)
#pragma unroll
    for (int s = 0; s < 8; ++s) {
      int m = s + 8 * g;
      O[m * 64 + t * 16 + r] = (*o[t])[s];
    }
  __syncthreads();

  // window part: accumulate 25-tap weighted v rows (lane owns 32 cols of query r)
  {
    int qi = qbase + r, rr = qi >> 5, cc = qi & 31;
    int cb = g * 32;
    float facc[32];
#pragma unroll
    for (int d = 0; d < 32; ++d) facc[d] = 0.f;
    for (int j = 0; j < WIN_; ++j) {
      int kr = rr + j / 5 - 2, kc = cc + j % 5 - 2;
      int kidx = kr * 32 + kc;
      if (kr < 0 || kr >= 32 || kc < 0 || kc >= 32 || kidx > qi) continue;
      float wgt = S[r * SS + 256 + j];
      const unsigned* v32 = (const unsigned*)(vb + ((size_t)head * SEQP_ + TXT_ + kidx) * DH_ + cb);
#pragma unroll
      for (int d = 0; d < 16; ++d) {
        unsigned uv = v32[d];
        facc[2 * d] += wgt * bf2f(uv & 0xffff);
        facc[2 * d + 1] += wgt * bf2f(uv >> 16);
      }
    }
#pragma unroll
    for (int d = 0; d < 32; ++d) O[r * 64 + cb + d] += facc[d];
  }
  __syncthreads();

  // store bf16 head-major output rows
  {
    int cb = g * 32;
    size_t ob = ((size_t)head * SEQP_ + TXT_ + qbase + r) * DH_ + cb;
#pragma unroll
    for (int d = 0; d < 32; ++d) ao[ob + d] = f2bf(O[r * 64 + cb + d]);
  }
}

// ---------------- output projection: (8,1280,512) @ w_out(512,512) + b ----------
// 32x64 output per wave (2 M x 4 N subtiles), same reuse scheme as k_qkv.
__global__ void __launch_bounds__(128) k_proj(const unsigned short* __restrict__ ao,
                                              const unsigned short* __restrict__ woutT,
                                              const float* __restrict__ b_out,
                                              float* __restrict__ out) {
  int wid = threadIdx.x >> 5, lane = threadIdx.x & 31;
  int w = blockIdx.x * 4 + wid;          // 8 * 40 * 8 waves
  int b = w / (40 * 8);
  int rem = w % (40 * 8);
  int mtg = rem >> 3, ng = rem & 7;
  int r = lane & 15, g = lane >> 4;
  int tr0 = mtg * 32 + r;
  int tr1 = tr0 + 16;

  const unsigned short* brow[4];
#pragma unroll
  for (int t = 0; t < 4; ++t)
    brow[t] = woutT + ((size_t)(ng * 64 + t * 16 + r)) * DIM_;

  f32x8 acc[2][4];
#pragma unroll
  for (int mi = 0; mi < 2; ++mi)
#pragma unroll
    for (int t = 0; t < 4; ++t) acc[mi][t] = zero8();

  for (int k0 = 0; k0 < DIM_; k0 += 32) {
    Frag a0, a1;
#pragma unroll
    for (int vi = 0; vi < 8; ++vi) {
      int k = k0 + kmap(vi, g);
      int h = k >> 6, d = k & 63;                // head-major gather, pairs share h
      size_t hb = (size_t)(b * 8 + h) * SEQP_;
      a0.u[vi] = *(const unsigned*)(ao + (hb + tr0) * DH_ + d);
      a1.u[vi] = *(const unsigned*)(ao + (hb + tr1) * DH_ + d);
    }
#pragma unroll
    for (int t = 0; t < 4; ++t) {
      Frag bf;
      load_frag_rows(bf, brow[t] + k0, g);
      acc[0][t] = wmma_bf16(a0, bf, acc[0][t]);
      acc[1][t] = wmma_bf16(a1, bf, acc[1][t]);
    }
  }
#pragma unroll
  for (int mi = 0; mi < 2; ++mi)
#pragma unroll
    for (int t = 0; t < 4; ++t) {
      int col = ng * 64 + t * 16 + r;
      float bias = b_out[col];
#pragma unroll
      for (int s = 0; s < 8; ++s) {
        int trow = mtg * 32 + mi * 16 + s + 8 * g;
        if (trow < SEQ_)
          out[((size_t)b * SEQ_ + trow) * DIM_ + col] = acc[mi][t][s] + bias;
      }
    }
}

// ---------------- host launcher ----------------
extern "C" void kernel_launch(void* const* d_in, const int* in_sizes, int n_in,
                              void* d_out, int out_size, void* d_ws, size_t ws_size,
                              hipStream_t stream) {
  const float* x = (const float*)d_in[0];
  const unsigned char* mask = (const unsigned char*)d_in[1];
  const float* w_qkv = (const float*)d_in[2];
  const float* w_out = (const float*)d_in[3];
  const float* b_out = (const float*)d_in[4];
  float* out = (float*)d_out;

  char* ws = (char*)d_ws;
  size_t off = 0;
  auto take = [&](size_t bytes) -> char* {
    char* p = ws + off;
    off = (off + bytes + 255) & ~(size_t)255;
    return p;
  };
  unsigned short* xb    = (unsigned short*)take((size_t)B_ * SEQP_ * DIM_ * 2);
  unsigned short* wqkvT = (unsigned short*)take((size_t)1536 * 512 * 2);
  unsigned short* woutT = (unsigned short*)take((size_t)512 * 512 * 2);
  unsigned short* qb    = (unsigned short*)take((size_t)NH_ * SEQP_ * DH_ * 2);
  unsigned short* kb    = (unsigned short*)take((size_t)NH_ * SEQP_ * DH_ * 2);
  unsigned short* vb    = (unsigned short*)take((size_t)NH_ * SEQP_ * DH_ * 2);
  unsigned short* vtt   = (unsigned short*)take((size_t)NH_ * DH_ * TXT_ * 2);
  unsigned short* ao    = (unsigned short*)take((size_t)NH_ * SEQP_ * DH_ * 2);

  k_cvt_x<<<(B_ * SEQP_ * DIM_ + 255) / 256, 256, 0, stream>>>(x, xb);
  k_cvt_wqkv<<<(1536 * 512 + 255) / 256, 256, 0, stream>>>(w_qkv, wqkvT);
  k_cvt_wout<<<(512 * 512 + 255) / 256, 256, 0, stream>>>(w_out, woutT);

  k_qkv<<<(B_ * 40 * 24) / 4, 128, 0, stream>>>(xb, wqkvT, qb, kb, vb, vtt);

  size_t txt_lds = 4u * 16 * 256 * sizeof(float);                 // 64 KB
  (void)hipFuncSetAttribute((const void*)k_text,
                            hipFuncAttributeMaxDynamicSharedMemorySize, (int)txt_lds);
  k_text<<<(NH_ * 16) / 4, 128, txt_lds, stream>>>(qb, kb, vtt, ao);

  size_t img_lds = 4u * (16 * 288 + 16 * 64) * sizeof(float);     // 88 KB
  (void)hipFuncSetAttribute((const void*)k_img,
                            hipFuncAttributeMaxDynamicSharedMemorySize, (int)img_lds);
  k_img<<<(NH_ * 64) / 4, 128, img_lds, stream>>>(qb, kb, vb, vtt, mask, ao);

  k_proj<<<(B_ * 40 * 8) / 4, 128, 0, stream>>>(ao, woutT, b_out, out);
}
